// SSSDBlock_22479858827558
// MI455X (gfx1250) — matrix operand order repaired and checked by reading, not compiled, not run on hardware
//
#include <hip/hip_runtime.h>
#include <math.h>

#define B_SZ 4
#define SEQ  2048
#define DM   256
#define DI   512
#define MTOT (B_SZ*SEQ)   /* 8192 */
#define NH   4
#define HD   64
#define DST  16

typedef __attribute__((ext_vector_type(16))) __bf16 v16bf;
typedef __attribute__((ext_vector_type(8)))  float  v8f;
typedef unsigned short u16;
typedef unsigned int   u32;

#define WMMA_BF16(a,b,c) __builtin_amdgcn_wmma_f32_16x16x32_bf16(false,(a),false,(b),(short)0,(c),false,false)

// CDNA5 async global->LDS copy path (guarded: falls back to load+ds_store)
#if defined(__has_builtin)
# if __has_builtin(__builtin_amdgcn_global_load_async_to_lds_b128) && __has_builtin(__builtin_amdgcn_s_wait_asynccnt)
#  define USE_ASYNC_LDS 1
# endif
#endif
#ifndef USE_ASYNC_LDS
# define USE_ASYNC_LDS 0
#endif

// Builtin prototype (from hipcc diagnostic): (V4i addrspace(1)*, V4i addrspace(3)*, Ii, Ii)
typedef int b128v __attribute__((vector_size(16)));

__device__ __forceinline__ u16 f2bf(float f) {
  u32 u = __float_as_uint(f);
  u32 r = u + 0x7FFFu + ((u >> 16) & 1u);
  return (u16)(r >> 16);
}
__device__ __forceinline__ float bf2f(u16 h) {
  return __uint_as_float(((u32)h) << 16);
}

union Frag16 { v16bf v; u16 s[16]; uint4 q[2]; };

// A-matrix fragment (16-bit, 16x32): lanes 0-15 hold K {0..15}, lanes 16-31 K {16..31}.
// p must already include the per-lane koff (= (lane>=16)?16:0). Chunks at p[0..7], p[8..15].
__device__ __forceinline__ v16bf load_frag_ilv(const u16* p) {
  Frag16 f;
  f.q[0] = *(const uint4*)(p);
  f.q[1] = *(const uint4*)(p + 8);
  return f.v;
}
// B-matrix fragment: per lane col = lane&15, K contiguous-16 per half. p includes half*16.
__device__ __forceinline__ v16bf load_frag_b(const u16* p) {
  Frag16 f;
  f.q[0] = *(const uint4*)(p);
  f.q[1] = *(const uint4*)(p + 8);
  return f.v;
}
__device__ __forceinline__ v8f vzero8() {
  v8f v;
#pragma unroll
  for (int i = 0; i < 8; ++i) v[i] = 0.f;
  return v;
}

// 16-byte global -> LDS copy (async when available)
__device__ __forceinline__ void copy16_g2l(const u16* g, u16* l) {
#if USE_ASYNC_LDS
  __builtin_amdgcn_global_load_async_to_lds_b128(
      (b128v __attribute__((address_space(1)))*)g,
      (b128v __attribute__((address_space(3)))*)l, 0, 0);
#else
  *(uint4*)l = *(const uint4*)g;
#endif
}

// ---------------- LayerNorm -> xn (bf16) ----------------
__global__ void __launch_bounds__(256) ln_kernel(const float* __restrict__ x,
                                                 const float* __restrict__ g,
                                                 const float* __restrict__ b,
                                                 u16* __restrict__ xn_bf) {
  __shared__ float red[256];
  const int row = blockIdx.x;
  const int tid = threadIdx.x;
  float v = x[(size_t)row * DM + tid];
  red[tid] = v; __syncthreads();
  for (int s = 128; s > 0; s >>= 1) { if (tid < s) red[tid] += red[tid + s]; __syncthreads(); }
  float mu = red[0] * (1.f / DM); __syncthreads();
  float d = v - mu;
  red[tid] = d * d; __syncthreads();
  for (int s = 128; s > 0; s >>= 1) { if (tid < s) red[tid] += red[tid + s]; __syncthreads(); }
  float var = red[0] * (1.f / DM);
  float xn = d * rsqrtf(var + 1e-5f) * g[tid] + b[tid];
  xn_bf[(size_t)row * DM + tid] = f2bf(xn);
}

// ---------------- weight convert + transpose: Wt[n*K+k] = bf16(W[k*N+n]) ----------------
__global__ void __launch_bounds__(256) wt_kernel(const float* __restrict__ W,
                                                 u16* __restrict__ Wt, int K, int N) {
  int idx = blockIdx.x * 256 + threadIdx.x;
  if (idx >= K * N) return;
  int n = idx % N, k = idx / N;
  Wt[(size_t)n * K + k] = f2bf(W[(size_t)k * N + n]);
}

// ---------------- bf16 WMMA GEMM ----------------
// 4-wave workgroup computes 64 x (NT*16) of C. B K-slice (NT*16 cols x 32 K) is staged
// in LDS (double-buffered, async copies) and shared by all 4 waves; A frags from global.
// epi: 0 store bf16; 1 store f32; 2 bias+bf16; 3 bias+f32; 4 final gate fusion (N==DM)
template<int NT>
__global__ void __launch_bounds__(128) gemm_bf16_wmma(
    const u16* __restrict__ A, const u16* __restrict__ Bt,
    int M, int N, int K, int epi,
    const float* __restrict__ bias,
    float* __restrict__ Cf, u16* __restrict__ Cb,
    const float* __restrict__ xres, const float* __restrict__ ssm,
    const float* __restrict__ attn, float* __restrict__ outf)
{
  __shared__ u16 bsh[2][64 * 32];           // [buf][n*32 + k], up to 4 KB per buffer
  const int tid  = threadIdx.x;
  const int lane = tid & 31;
  const int wave = tid >> 5;
  const int half = lane >> 4;
  const int rA   = lane & 15;
  const int koff = half * 16;
  const int tm   = blockIdx.y * 64 + wave * 16;
  const int tn0  = blockIdx.x * (NT * 16);

  // stage one 32-K slice of the B panel into bsh[buf]; 16B chunk per thread slot
  auto stage = [&](int buf, int kb) {
    {
      int c = tid;                          // c < NT*64 always (NT >= 3 -> 192 > 127)
      int n = c >> 2, kc = (c & 3) * 8;
      copy16_g2l(Bt + (size_t)(tn0 + n) * K + kb + kc, &bsh[buf][n * 32 + kc]);
    }
    int c = tid + 128;
    if (c < NT * 64) {
      int n = c >> 2, kc = (c & 3) * 8;
      copy16_g2l(Bt + (size_t)(tn0 + n) * K + kb + kc, &bsh[buf][n * 32 + kc]);
    }
  };

  v8f acc[NT];
#pragma unroll
  for (int t = 0; t < NT; ++t) acc[t] = vzero8();

  const u16* arow = A + (size_t)(tm + rA) * K;

  stage(0, 0);
  for (int kb = 0; kb < K; kb += 32) {
    const int cur = (kb >> 5) & 1;
#if USE_ASYNC_LDS
    __builtin_amdgcn_s_wait_asynccnt(0);
#endif
    __syncthreads();                        // staged slice visible to all 4 waves
    if (kb + 32 < K) stage(cur ^ 1, kb + 32);     // overlap next copy with compute
    if (kb + 64 < K) __builtin_prefetch(arow + kb + 64, 0, 1);
    v16bf af = load_frag_ilv(arow + kb + koff);
#pragma unroll
    for (int t = 0; t < NT; ++t) {
      v16bf bfv = load_frag_b(&bsh[cur][(t * 16 + rA) * 32 + half * 16]);
      acc[t] = WMMA_BF16(af, bfv, acc[t]);
    }
  }

#pragma unroll
  for (int t = 0; t < NT; ++t) {
    int col = tn0 + t * 16 + rA;
#pragma unroll
    for (int r = 0; r < 8; ++r) {
      int row = tm + r + half * 8;
      float v = acc[t][r];
      size_t idx = (size_t)row * N + col;
      if (epi == 0)      Cb[idx] = f2bf(v);
      else if (epi == 1) Cf[idx] = v;
      else if (epi == 2) Cb[idx] = f2bf(v + bias[col]);
      else if (epi == 3) Cf[idx] = v + bias[col];
      else {
        float gsig = 1.f / (1.f + __expf(-(v + bias[col])));
        size_t di = (size_t)row * DM + col;
        outf[di] = xres[di] + gsig * ssm[di] + (1.f - gsig) * attn[di];
      }
    }
  }
}

// ---------------- depthwise conv (k=4) + SiLU ----------------
__global__ void __launch_bounds__(256) conv_kernel(const u16* __restrict__ xz_bf,
                                                   const float* __restrict__ conv_w,
                                                   const float* __restrict__ conv_b,
                                                   float* __restrict__ xc_f,
                                                   u16* __restrict__ xc_bf) {
  int idx = blockIdx.x * 256 + threadIdx.x;
  if (idx >= MTOT * DI) return;
  int d = idx & (DI - 1);
  int m = idx >> 9;
  int b = m >> 11, t = m & (SEQ - 1);
  float acc = conv_b[d];
#pragma unroll
  for (int k = 0; k < 4; ++k) {
    int tt = t - 3 + k;
    if (tt >= 0)
      acc += bf2f(xz_bf[((size_t)(b * SEQ + tt)) * (2 * DI) + d]) * conv_w[d * 4 + k];
  }
  float s = acc / (1.f + __expf(-acc));
  xc_f[idx] = s;
  xc_bf[idx] = f2bf(s);
}

// ---------------- dt = softplus(dt_in @ W_dt + b_dt), K=16 scalar ----------------
__global__ void __launch_bounds__(256) dt_kernel(const float* __restrict__ dbl,
                                                 const float* __restrict__ W_dt,
                                                 const float* __restrict__ b_dt,
                                                 float* __restrict__ dtb) {
  int idx = blockIdx.x * 256 + threadIdx.x;
  if (idx >= MTOT * DI) return;
  int d = idx & (DI - 1);
  int m = idx >> 9;
  float acc = b_dt[d];
#pragma unroll
  for (int r = 0; r < 16; ++r)
    acc += dbl[(size_t)m * 48 + r] * W_dt[r * DI + d];
  float sp = (acc > 20.f) ? acc : log1pf(__expf(acc));
  dtb[idx] = sp;
}

// ---------------- selective-scan: one thread per (b, d), 16 states in regs ----------------
__global__ void __launch_bounds__(256) scan_kernel(const float* __restrict__ xc,
                                                   const float* __restrict__ dtb,
                                                   const float* __restrict__ dbl,
                                                   const u16* __restrict__ xz_bf,
                                                   const float* __restrict__ A_log,
                                                   const float* __restrict__ Dp,
                                                   u16* __restrict__ y_bf) {
  int g = blockIdx.x * 256 + threadIdx.x;
  if (g >= B_SZ * DI) return;
  int d = g & (DI - 1);
  int b = g >> 9;
  float Ar[DST], h[DST];
#pragma unroll
  for (int n = 0; n < DST; ++n) { Ar[n] = -__expf(A_log[d * DST + n]); h[n] = 0.f; }
  float dp = Dp[d];
  for (int t = 0; t < SEQ; ++t) {
    size_t base = (size_t)(b * SEQ + t);
    float xt  = xc [base * DI + d];
    float dtv = dtb[base * DI + d];
    const float* bc = dbl + base * 48;
    float dx = dtv * xt;
    float y = 0.f;
#pragma unroll
    for (int n = 0; n < DST; ++n) {
      float dA = __expf(dtv * Ar[n]);
      h[n] = dA * h[n] + dx * bc[16 + n];
      y += h[n] * bc[32 + n];
    }
    float z = bf2f(xz_bf[base * (2 * DI) + DI + d]);
    float sz = z / (1.f + __expf(-z));
    y_bf[base * DI + d] = f2bf((y + xt * dp) * sz);
  }
}

// ---------------- V transpose: vt[(bh*64+d)*L + t] = v[b,t,h,d] ----------------
__global__ void __launch_bounds__(256) vt_kernel(const u16* __restrict__ qkv,
                                                 u16* __restrict__ vt) {
  int idx = blockIdx.x * 256 + threadIdx.x;
  if (idx >= B_SZ * NH * HD * SEQ) return;
  int t  = idx & (SEQ - 1);
  int dd = (idx >> 11) & 63;
  int h  = (idx >> 17) & 3;
  int b  = idx >> 19;
  vt[idx] = qkv[((size_t)(b * SEQ + t)) * (3 * DM) + 2 * DM + h * HD + dd];
}

// ---------------- flash attention: one wave per (b,h,q-tile) ----------------
__global__ void __launch_bounds__(32) attn_wmma(const u16* __restrict__ qkv,
                                                const u16* __restrict__ vt,
                                                u16* __restrict__ omerged) {
  __shared__ u16 pbuf[16 * 32];
  const int lane = threadIdx.x & 31;
  const int half = lane >> 4;
  const int rA   = lane & 15;
  const int koff = half * 16;
  const int qt   = blockIdx.x;
  const int bh   = blockIdx.y;
  const int b    = bh >> 2, h = bh & 3;

  const u16* qrow = qkv + ((size_t)(b * SEQ + qt * 16 + rA)) * (3 * DM) + h * HD;
  v16bf qf0 = load_frag_ilv(qrow + 0  + koff);
  v16bf qf1 = load_frag_ilv(qrow + 32 + koff);

  v8f o[4];
#pragma unroll
  for (int t = 0; t < 4; ++t) o[t] = vzero8();
  float mrow[8], lrow[8];
#pragma unroll
  for (int r = 0; r < 8; ++r) { mrow[r] = -1e30f; lrow[r] = 0.f; }
  const float scale = 0.125f; // 1/sqrt(64)

  for (int j = 0; j < SEQ; j += 32) {
    v8f s0 = vzero8(), s1 = vzero8();
    const u16* k0 = qkv + ((size_t)(b * SEQ + j + rA))      * (3 * DM) + DM + h * HD;
    const u16* k1 = qkv + ((size_t)(b * SEQ + j + 16 + rA)) * (3 * DM) + DM + h * HD;
    {
      v16bf kf;
      kf = load_frag_b(k0 + 0  + half * 16); s0 = WMMA_BF16(qf0, kf, s0);
      kf = load_frag_b(k0 + 32 + half * 16); s0 = WMMA_BF16(qf1, kf, s0);
      kf = load_frag_b(k1 + 0  + half * 16); s1 = WMMA_BF16(qf0, kf, s1);
      kf = load_frag_b(k1 + 32 + half * 16); s1 = WMMA_BF16(qf1, kf, s1);
    }
    float fac[8];
#pragma unroll
    for (int r = 0; r < 8; ++r) {
      float a = s0[r] * scale, c = s1[r] * scale;
      float mv = fmaxf(a, c);
#pragma unroll
      for (int msk = 1; msk <= 8; msk <<= 1) mv = fmaxf(mv, __shfl_xor(mv, msk, 32));
      float mnew = fmaxf(mrow[r], mv);
      float f = __expf(mrow[r] - mnew);
      float p0 = __expf(a - mnew);
      float p1 = __expf(c - mnew);
      float rs = p0 + p1;
#pragma unroll
      for (int msk = 1; msk <= 8; msk <<= 1) rs += __shfl_xor(rs, msk, 32);
      lrow[r] = lrow[r] * f + rs;
      mrow[r] = mnew;
      fac[r] = f;
      s0[r] = p0; s1[r] = p1;
    }
#pragma unroll
    for (int t = 0; t < 4; ++t)
#pragma unroll
      for (int r = 0; r < 8; ++r) o[t][r] *= fac[r];

    // stage P (16x32, C-layout) through LDS, re-read in A-fragment layout
    __syncthreads();
#pragma unroll
    for (int r = 0; r < 8; ++r) {
      int row = r + half * 8;
      pbuf[row * 32 + rA]      = f2bf(s0[r]);
      pbuf[row * 32 + 16 + rA] = f2bf(s1[r]);
    }
    __syncthreads();
    v16bf pf = load_frag_ilv(&pbuf[rA * 32 + koff]);
#pragma unroll
    for (int t = 0; t < 4; ++t) {
      const u16* vrow = vt + ((size_t)(bh * HD + t * 16 + rA)) * SEQ + j + half * 16;
      v16bf vf = load_frag_b(vrow);
      o[t] = WMMA_BF16(pf, vf, o[t]);
    }
  }
#pragma unroll
  for (int t = 0; t < 4; ++t)
#pragma unroll
    for (int r = 0; r < 8; ++r) {
      int row = qt * 16 + r + half * 8;
      float v = o[t][r] / lrow[r];
      omerged[((size_t)(b * SEQ + row)) * DM + h * HD + t * 16 + rA] = f2bf(v);
    }
}

// ---------------- pack concat(ssm, attn) -> bf16 ----------------
__global__ void __launch_bounds__(256) pack_kernel(const float* __restrict__ ssm,
                                                   const float* __restrict__ attn,
                                                   u16* __restrict__ cat) {
  int idx = blockIdx.x * 256 + threadIdx.x;
  if (idx >= MTOT * 2 * DM) return;
  int c = idx & (2 * DM - 1);
  int m = idx >> 9;
  float v = (c < DM) ? ssm[(size_t)m * DM + c] : attn[(size_t)m * DM + c - DM];
  cat[idx] = f2bf(v);
}

extern "C" void kernel_launch(void* const* d_in, const int* in_sizes, int n_in,
                              void* d_out, int out_size, void* d_ws, size_t ws_size,
                              hipStream_t stream) {
  (void)in_sizes; (void)n_in; (void)out_size; (void)ws_size;
  const float* x      = (const float*)d_in[0];
  const float* ln_g   = (const float*)d_in[1];
  const float* ln_b   = (const float*)d_in[2];
  const float* W_in   = (const float*)d_in[3];
  const float* conv_w = (const float*)d_in[4];
  const float* conv_b = (const float*)d_in[5];
  const float* W_xproj= (const float*)d_in[6];
  const float* W_dt   = (const float*)d_in[7];
  const float* b_dt   = (const float*)d_in[8];
  const float* A_log  = (const float*)d_in[9];
  const float* Dp     = (const float*)d_in[10];
  const float* W_out_m= (const float*)d_in[11];
  const float* W_qkv  = (const float*)d_in[12];
  const float* b_qkv  = (const float*)d_in[13];
  const float* W_o    = (const float*)d_in[14];
  const float* b_o    = (const float*)d_in[15];
  const float* W_gate = (const float*)d_in[16];
  const float* b_gate = (const float*)d_in[17];
  float* out = (float*)d_out;

  char* w = (char*)d_ws;
  size_t off = 0;
  auto alloc = [&](size_t bytes) -> void* {
    void* p = w + off;
    off = (off + bytes + 255) & ~(size_t)255;
    return p;
  };
  u16*   xn_bf   = (u16*)  alloc((size_t)MTOT * DM * 2);
  u16*   xz_bf   = (u16*)  alloc((size_t)MTOT * 2 * DI * 2);
  float* xc_f    = (float*)alloc((size_t)MTOT * DI * 4);
  u16*   xc_bf   = (u16*)  alloc((size_t)MTOT * DI * 2);
  float* dbl     = (float*)alloc((size_t)MTOT * 48 * 4);
  float* dtb     = (float*)alloc((size_t)MTOT * DI * 4);
  u16*   y_bf    = (u16*)  alloc((size_t)MTOT * DI * 2);
  float* ssm     = (float*)alloc((size_t)MTOT * DM * 4);
  u16*   qkv_bf  = (u16*)  alloc((size_t)MTOT * 3 * DM * 2);
  u16*   vt      = (u16*)  alloc((size_t)B_SZ * NH * HD * SEQ * 2);
  u16*   o_bf    = (u16*)  alloc((size_t)MTOT * DM * 2);
  float* attn    = (float*)alloc((size_t)MTOT * DM * 4);
  u16*   cat     = (u16*)  alloc((size_t)MTOT * 2 * DM * 2);
  u16*   W_in_t  = (u16*)  alloc((size_t)DM * 2 * DI * 2);
  u16*   W_xp_t  = (u16*)  alloc((size_t)DI * 48 * 2);
  u16*   W_om_t  = (u16*)  alloc((size_t)DI * DM * 2);
  u16*   W_qkv_t = (u16*)  alloc((size_t)DM * 3 * DM * 2);
  u16*   W_o_t   = (u16*)  alloc((size_t)DM * DM * 2);
  u16*   W_g_t   = (u16*)  alloc((size_t)(2 * DM) * DM * 2);

  auto cvt = [&](const float* W, u16* Wt, int K, int N) {
    int tot = K * N;
    wt_kernel<<<dim3((tot + 255) / 256), dim3(256), 0, stream>>>(W, Wt, K, N);
  };
  auto gemm = [&](const u16* A, const u16* Bt, int M, int N, int K, int epi,
                  const float* bias, float* Cf, u16* Cb,
                  const float* xr, const float* sp, const float* ap, float* op) {
    if (N % 64 == 0) {
      dim3 grid(N / 64, M / 64);
      gemm_bf16_wmma<4><<<grid, dim3(128), 0, stream>>>(A, Bt, M, N, K, epi, bias, Cf, Cb, xr, sp, ap, op);
    } else { // N == 48
      dim3 grid(1, M / 64);
      gemm_bf16_wmma<3><<<grid, dim3(128), 0, stream>>>(A, Bt, M, N, K, epi, bias, Cf, Cb, xr, sp, ap, op);
    }
  };

  // weights -> transposed bf16 (live entirely in L2)
  cvt(W_in,    W_in_t,  DM, 2 * DI);
  cvt(W_xproj, W_xp_t,  DI, 48);
  cvt(W_out_m, W_om_t,  DI, DM);
  cvt(W_qkv,   W_qkv_t, DM, 3 * DM);
  cvt(W_o,     W_o_t,   DM, DM);
  cvt(W_gate,  W_g_t,   2 * DM, DM);

  // 1. layernorm
  ln_kernel<<<dim3(MTOT), dim3(256), 0, stream>>>(x, ln_g, ln_b, xn_bf);
  // 2. xz = xn @ W_in (-> bf16)
  gemm(xn_bf, W_in_t, MTOT, 2 * DI, DM, 0, nullptr, nullptr, xz_bf, nullptr, nullptr, nullptr, nullptr);
  // 3. depthwise conv + silu
  conv_kernel<<<dim3((MTOT * DI + 255) / 256), dim3(256), 0, stream>>>(xz_bf, conv_w, conv_b, xc_f, xc_bf);
  // 4. dbl = xc @ W_xproj (-> f32)
  gemm(xc_bf, W_xp_t, MTOT, 48, DI, 1, nullptr, dbl, nullptr, nullptr, nullptr, nullptr, nullptr);
  // 5. dt = softplus(dt_in @ W_dt + b_dt)
  dt_kernel<<<dim3((MTOT * DI + 255) / 256), dim3(256), 0, stream>>>(dbl, W_dt, b_dt, dtb);
  // 6. selective scan (+ Dp skip + silu(z) gate) -> y bf16
  scan_kernel<<<dim3((B_SZ * DI + 255) / 256), dim3(256), 0, stream>>>(xc_f, dtb, dbl, xz_bf, A_log, Dp, y_bf);
  // 7. ssm_out = y @ W_out_m (-> f32)
  gemm(y_bf, W_om_t, MTOT, DM, DI, 1, nullptr, ssm, nullptr, nullptr, nullptr, nullptr, nullptr);
  // 8. qkv = xn @ W_qkv + b_qkv (-> bf16)
  gemm(xn_bf, W_qkv_t, MTOT, 3 * DM, DM, 2, b_qkv, nullptr, qkv_bf, nullptr, nullptr, nullptr, nullptr);
  // 9. transpose V per head
  vt_kernel<<<dim3((B_SZ * NH * HD * SEQ + 255) / 256), dim3(256), 0, stream>>>(qkv_bf, vt);
  // 10. flash attention -> merged (B,L,256) bf16
  attn_wmma<<<dim3(SEQ / 16, B_SZ * NH), dim3(32), 0, stream>>>(qkv_bf, vt, o_bf);
  // 11. attn_out = o @ W_o + b_o (-> f32)
  gemm(o_bf, W_o_t, MTOT, DM, DM, 3, b_o, attn, nullptr, nullptr, nullptr, nullptr, nullptr);
  // 12. pack concat(ssm, attn) -> bf16
  pack_kernel<<<dim3((MTOT * 2 * DM + 255) / 256), dim3(256), 0, stream>>>(ssm, attn, cat);
  // 13. gate GEMM with fused sigmoid-gate + residual -> d_out
  gemm(cat, W_g_t, MTOT, DM, 2 * DM, 4, b_gate, nullptr, nullptr, x, ssm, attn, out);
}
